// AttnBlock_62182536512160
// MI455X (gfx1250) — matrix operand order repaired and checked by reading, not compiled, not run on hardware
//
#include <hip/hip_runtime.h>
#include <hip/hip_bf16.h>

// ---------------------------------------------------------------------------
// AttnBlock for MI455X (gfx1250): GroupNorm -> QKV proj -> full attention ->
// proj + residual.  All heavy math via v_wmma_f32_16x16x32_bf16 (wave32).
// ---------------------------------------------------------------------------

typedef __attribute__((ext_vector_type(16))) __bf16 v16bf;
typedef __attribute__((ext_vector_type(8)))  float  v8f;

constexpr int   CCH    = 512;   // channels
constexpr int   SSP    = 1024;  // H*W
constexpr int   NBATCH = 32;
constexpr int   NGRP   = 32;
constexpr int   CPG    = CCH / NGRP;  // 16
constexpr float GN_EPS = 1e-6f;

constexpr int BM = 128, BN = 128, BK = 32, BKP = 40;  // BKP*2B = 80B rows (16B aligned)
constexpr int NTHREADS = 256;                          // 8 waves (wave32)

struct alignas(16) U128 { unsigned int w0, w1, w2, w3; };
union Frag {
  U128   u[2];
  v16bf  v;
  __bf16 e[16];
};

// ---------------------------------------------------------------------------
// GroupNorm: one workgroup per (batch, group); group = 16 contiguous channel
// rows of 1024 = 16384 contiguous floats. Writes bf16 activations.
// ---------------------------------------------------------------------------
__global__ __launch_bounds__(NTHREADS)
void gn_kernel(const float* __restrict__ x, const float* __restrict__ gamma,
               const float* __restrict__ beta, __bf16* __restrict__ h)
{
  __shared__ float r0[NTHREADS], r1[NTHREADS];
  const int b = blockIdx.x / NGRP;
  const int g = blockIdx.x % NGRP;
  const long base = ((long)b * CCH + (long)g * CPG) * SSP;
  const float* xp = x + base;
  const int n = CPG * SSP;  // 16384
  float s0 = 0.f, s1 = 0.f;
  for (int i = threadIdx.x; i < n; i += NTHREADS) {
    float v = xp[i]; s0 += v; s1 += v * v;
  }
  r0[threadIdx.x] = s0; r1[threadIdx.x] = s1;
  __syncthreads();
  for (int off = NTHREADS / 2; off > 0; off >>= 1) {
    if (threadIdx.x < off) {
      r0[threadIdx.x] += r0[threadIdx.x + off];
      r1[threadIdx.x] += r1[threadIdx.x + off];
    }
    __syncthreads();
  }
  const float inv_n = 1.0f / (float)n;
  const float mean  = r0[0] * inv_n;
  const float var   = r1[0] * inv_n - mean * mean;
  const float rstd  = rsqrtf(var + GN_EPS);
  __bf16* hp = h + base;
  for (int i = threadIdx.x; i < n; i += NTHREADS) {
    const int c = g * CPG + (i >> 10);  // i / 1024
    const float v = (xp[i] - mean) * rstd * gamma[c] + beta[c];
    hp[i] = (__bf16)v;
  }
}

// ---------------------------------------------------------------------------
// Transpose the four 512x512 fp32 weights into bf16 (d, c) layout so the GEMM
// A-tile is K-contiguous. Tiny (1 MB reads each), L2-absorbed.
// ---------------------------------------------------------------------------
__global__ __launch_bounds__(NTHREADS)
void wtrans_kernel(const float* __restrict__ Wq, const float* __restrict__ Wk,
                   const float* __restrict__ Wv, const float* __restrict__ Wp,
                   __bf16* __restrict__ wt)
{
  const long t = (long)blockIdx.x * NTHREADS + threadIdx.x;
  const int m = (int)(t >> 18);       // which matrix (262144 elems each)
  const int i = (int)(t & 262143);
  const float* Ws = (m == 0) ? Wq : (m == 1) ? Wk : (m == 2) ? Wv : Wp;
  const int d = i >> 9, c = i & 511;
  wt[t] = (__bf16)Ws[c * 512 + d];    // wt[m][d][c] = W[c][d]
}

// ---------------------------------------------------------------------------
// Row softmax over 1024 bf16 scores, fp32 math, in place.
// ---------------------------------------------------------------------------
__global__ __launch_bounds__(NTHREADS)
void softmax_kernel(__bf16* __restrict__ w)
{
  __shared__ float red[NTHREADS];
  __bf16* p = w + (long)blockIdx.x * SSP;
  float v[4];
  float mx = -3.0e38f;
#pragma unroll
  for (int j = 0; j < 4; ++j) {
    v[j] = (float)p[threadIdx.x + j * NTHREADS];
    mx = fmaxf(mx, v[j]);
  }
  red[threadIdx.x] = mx; __syncthreads();
  for (int off = NTHREADS / 2; off > 0; off >>= 1) {
    if (threadIdx.x < off) red[threadIdx.x] = fmaxf(red[threadIdx.x], red[threadIdx.x + off]);
    __syncthreads();
  }
  mx = red[0]; __syncthreads();
  float s = 0.f;
#pragma unroll
  for (int j = 0; j < 4; ++j) { v[j] = __expf(v[j] - mx); s += v[j]; }
  red[threadIdx.x] = s; __syncthreads();
  for (int off = NTHREADS / 2; off > 0; off >>= 1) {
    if (threadIdx.x < off) red[threadIdx.x] += red[threadIdx.x + off];
    __syncthreads();
  }
  const float inv = 1.0f / red[0];
#pragma unroll
  for (int j = 0; j < 4; ++j) p[threadIdx.x + j * NTHREADS] = (__bf16)(v[j] * inv);
}

// ---------------------------------------------------------------------------
// Tiled bf16 WMMA GEMM: C[M,N] = A[M,K] * B[K,N] per batch z.
//   AT: A stored (K,M) -> transpose while filling LDS.   !AT: (M,K) direct.
//   BT: B stored (K,N) -> transpose while filling LDS.   !BT: (N,K) direct.
// LDS holds A as [m][k] and B as [n][k] (K-contiguous, BKP=40 pad) so every
// WMMA fragment is exactly two aligned ds_load_b128 in contiguous k order (same for A and B):
//   A frag : lane L row m=L%16, halves 0-15 -> K=kb..kb+15,
//            kb = 16*(L>>4)
//   B frag : lane L col n=L%16, halves 0-15 -> K=kb..kb+15, kb = 16*(L>>4)
// 8 waves in 2(M)x4(N); each wave owns a 64x32 tile = 4x2 f32 accumulators.
// Double-buffered LDS, software pipelined, one barrier per K step.
// EPI: 0 = +bias -> bf16 | 1 = *scale -> bf16 | 2 = bf16 | 3 = +bias+resid -> f32
// ---------------------------------------------------------------------------
template<bool AT, bool BT, int EPI>
__global__ __launch_bounds__(NTHREADS)
void gemm_bf16(const __bf16* __restrict__ Ag, const __bf16* __restrict__ Bg,
               void* __restrict__ Og,
               const float* __restrict__ bias, const float* __restrict__ resid,
               int K, int lda, int ldb, int ldo,
               long strideA, long strideB, long strideO, long strideR,
               float scale)
{
  __shared__ alignas(16) __bf16 smem[2 * (BM + BN) * BKP];  // 40 KB
  const int tid  = threadIdx.x;
  const int lane = tid & 31;
  const int wav  = tid >> 5;
  const int wrow = wav & 1;   // 2 waves along M
  const int wcol = wav >> 1;  // 4 waves along N
  const int tileN = blockIdx.x * BN;
  const int tileM = blockIdx.y * BM;
  const long bz = blockIdx.z;
  Ag += bz * strideA;
  Bg += bz * strideB;

  const int KT = K / BK;

  U128 ra0, ra1, rb0, rb1;

  auto gload = [&](int kt) {
    if (!AT) {  // A (M,K): 2 threads per row, 16 halves each (32B)
      const int r = tid >> 1, kk = (tid & 1) * 16;
      const U128* p = (const U128*)(Ag + (long)(tileM + r) * lda + kt * BK + kk);
      ra0 = p[0]; ra1 = p[1];
    } else {    // A (K,M): lane-distinct k rows, 16 contiguous m each
      const int k = tid & 31, n0 = (tid >> 5) * 16;
      const U128* p = (const U128*)(Ag + (long)(kt * BK + k) * lda + tileM + n0);
      ra0 = p[0]; ra1 = p[1];
    }
    if (!BT) {
      const int r = tid >> 1, kk = (tid & 1) * 16;
      const U128* p = (const U128*)(Bg + (long)(tileN + r) * ldb + kt * BK + kk);
      rb0 = p[0]; rb1 = p[1];
    } else {
      const int k = tid & 31, n0 = (tid >> 5) * 16;
      const U128* p = (const U128*)(Bg + (long)(kt * BK + k) * ldb + tileN + n0);
      rb0 = p[0]; rb1 = p[1];
    }
  };

  auto sstore = [&](int buf) {
    __bf16* As = smem + buf * (BM + BN) * BKP;
    __bf16* Bs = As + BM * BKP;
    if (!AT) {
      const int r = tid >> 1, kk = (tid & 1) * 16;
      *(U128*)(As + r * BKP + kk)     = ra0;
      *(U128*)(As + r * BKP + kk + 8) = ra1;
    } else {
      const int k = tid & 31, n0 = (tid >> 5) * 16;
      Frag f; f.u[0] = ra0; f.u[1] = ra1;
#pragma unroll
      for (int i = 0; i < 16; ++i) As[(n0 + i) * BKP + k] = f.e[i];
    }
    if (!BT) {
      const int r = tid >> 1, kk = (tid & 1) * 16;
      *(U128*)(Bs + r * BKP + kk)     = rb0;
      *(U128*)(Bs + r * BKP + kk + 8) = rb1;
    } else {
      const int k = tid & 31, n0 = (tid >> 5) * 16;
      Frag f; f.u[0] = rb0; f.u[1] = rb1;
#pragma unroll
      for (int i = 0; i < 16; ++i) Bs[(n0 + i) * BKP + k] = f.e[i];
    }
  };

  v8f acc[4][2];
#pragma unroll
  for (int a = 0; a < 4; ++a)
#pragma unroll
    for (int c = 0; c < 2; ++c)
#pragma unroll
      for (int j = 0; j < 8; ++j) acc[a][c][j] = 0.0f;

  gload(0);
  sstore(0);
  __syncthreads();

  const int kbA = (lane >> 4) * 16;
  const int kbB = (lane >> 4) * 16;
  const int lm  = lane & 15;

  for (int kt = 0; kt < KT; ++kt) {
    const int  cur  = kt & 1;
    const bool more = (kt + 1) < KT;
    if (more) gload(kt + 1);  // overlap global latency with WMMA

    const __bf16* As = smem + cur * (BM + BN) * BKP;
    const __bf16* Bs = As + BM * BKP;

#pragma unroll
    for (int fn = 0; fn < 2; ++fn) {
      Frag bf;
      const int n = wcol * 32 + fn * 16 + lm;
      bf.u[0] = *(const U128*)(Bs + n * BKP + kbB);
      bf.u[1] = *(const U128*)(Bs + n * BKP + kbB + 8);
#pragma unroll
      for (int fm = 0; fm < 4; ++fm) {
        Frag af;
        const int m = wrow * 64 + fm * 16 + lm;
        af.u[0] = *(const U128*)(As + m * BKP + kbA);
        af.u[1] = *(const U128*)(As + m * BKP + kbA + 8);
        acc[fm][fn] = __builtin_amdgcn_wmma_f32_16x16x32_bf16(
            false, af.v, false, bf.v, (short)0, acc[fm][fn], false, false);
      }
    }
    if (more) sstore(1 - cur);  // safe: buf(1-cur) last read in kt-1 (barrier)
    __syncthreads();
  }

  // Epilogue. C/D layout: lane L -> n = L%16; VGPR j -> m = j + 8*(L>>4).
  const int mBase = tileM + wrow * 64 + ((lane >> 4) << 3);
  const int nBase = tileN + wcol * 32 + lm;
#pragma unroll
  for (int fm = 0; fm < 4; ++fm) {
#pragma unroll
    for (int fn = 0; fn < 2; ++fn) {
#pragma unroll
      for (int j = 0; j < 8; ++j) {
        const int m = mBase + fm * 16 + j;
        const int n = nBase + fn * 16;
        float vv = acc[fm][fn][j];
        const long oidx = bz * strideO + (long)m * ldo + n;
        if (EPI == 0) {
          ((__bf16*)Og)[oidx] = (__bf16)(vv + bias[m]);
        } else if (EPI == 1) {
          ((__bf16*)Og)[oidx] = (__bf16)(vv * scale);
        } else if (EPI == 2) {
          ((__bf16*)Og)[oidx] = (__bf16)vv;
        } else {
          vv += bias[m] + resid[bz * strideR + (long)m * ldo + n];
          ((float*)Og)[oidx] = vv;
        }
      }
    }
  }
}

// ---------------------------------------------------------------------------
// Orchestration. Workspace (~237 MB):
//   h, q, k, v : bf16 (B, C, S)  32 MB each
//   w          : bf16 (B, S, S)  64 MB (scores; softmax in place)
//   h_out      : bf16 (B, C, S)  32 MB
//   wt         : bf16 4x(512,512) transposed weights, 2 MB
// ---------------------------------------------------------------------------
extern "C" void kernel_launch(void* const* d_in, const int* in_sizes, int n_in,
                              void* d_out, int out_size, void* d_ws, size_t ws_size,
                              hipStream_t stream)
{
  (void)in_sizes; (void)n_in; (void)out_size; (void)ws_size;
  const float* x     = (const float*)d_in[0];
  const float* gamma = (const float*)d_in[1];
  const float* beta  = (const float*)d_in[2];
  const float* Wq    = (const float*)d_in[3];
  const float* bq    = (const float*)d_in[4];
  const float* Wk    = (const float*)d_in[5];
  const float* bk    = (const float*)d_in[6];
  const float* Wv    = (const float*)d_in[7];
  const float* bv    = (const float*)d_in[8];
  const float* Wp    = (const float*)d_in[9];
  const float* bp    = (const float*)d_in[10];

  const long CS = (long)CCH * SSP;      // 524288 elems per batch (C,S)
  const long SS = (long)SSP * SSP;      // 1048576 elems per batch (S,S)
  const long szCS = 2L * CS * NBATCH;   // 32 MB
  const long szSS = 2L * SS * NBATCH;   // 64 MB

  char* ws = (char*)d_ws;
  __bf16* h  = (__bf16*)(ws);
  __bf16* q  = (__bf16*)(ws + 1 * szCS);
  __bf16* k_ = (__bf16*)(ws + 2 * szCS);
  __bf16* v_ = (__bf16*)(ws + 3 * szCS);
  __bf16* w_ = (__bf16*)(ws + 4 * szCS);
  __bf16* ho = (__bf16*)(ws + 4 * szCS + szSS);
  __bf16* wt = (__bf16*)(ws + 5 * szCS + szSS);

  wtrans_kernel<<<4096, NTHREADS, 0, stream>>>(Wq, Wk, Wv, Wp, wt);
  gn_kernel<<<NBATCH * NGRP, NTHREADS, 0, stream>>>(x, gamma, beta, h);

  const dim3 gP(SSP / BN, CCH / BM, NBATCH);  // (8,4,32) projections
  const dim3 gS(SSP / BN, SSP / BM, NBATCH);  // (8,8,32) scores
  const __bf16* Wtq = wt;
  const __bf16* Wtk = wt + 1L * 262144;
  const __bf16* Wtv = wt + 2L * 262144;
  const __bf16* Wtp = wt + 3L * 262144;

  // q/k/v[d,s] = sum_c Wt[d,c] * h[c,s] + b[d]
  gemm_bf16<false, true, 0><<<gP, NTHREADS, 0, stream>>>(
      Wtq, h, q, bq, nullptr, 512, 512, 1024, 1024, 0, CS, CS, 0, 1.0f);
  gemm_bf16<false, true, 0><<<gP, NTHREADS, 0, stream>>>(
      Wtk, h, k_, bk, nullptr, 512, 512, 1024, 1024, 0, CS, CS, 0, 1.0f);
  gemm_bf16<false, true, 0><<<gP, NTHREADS, 0, stream>>>(
      Wtv, h, v_, bv, nullptr, 512, 512, 1024, 1024, 0, CS, CS, 0, 1.0f);

  // w[s,t] = (sum_c q[c,s] * k[c,t]) * C^-0.5
  gemm_bf16<true, true, 1><<<gS, NTHREADS, 0, stream>>>(
      q, k_, w_, nullptr, nullptr, 512, 1024, 1024, 1024, CS, CS, SS, 0,
      0.04419417382415922f);

  softmax_kernel<<<NBATCH * SSP, NTHREADS, 0, stream>>>(w_);

  // h_out[c,s] = sum_t v[c,t] * w[s,t]
  gemm_bf16<false, false, 2><<<gP, NTHREADS, 0, stream>>>(
      v_, w_, ho, nullptr, nullptr, 1024, 1024, 1024, 1024, CS, SS, CS, 0, 1.0f);

  // out[d,s] = x[d,s] + sum_c Wpt[d,c] * h_out[c,s] + bp[d]   (fp32)
  gemm_bf16<false, true, 3><<<gP, NTHREADS, 0, stream>>>(
      Wtp, ho, d_out, bp, x, 512, 512, 1024, 1024, 0, CS, CS, CS, 1.0f);
}
